// MultiHeadedAttention_764504179129
// MI455X (gfx1250) — matrix operand order repaired and checked by reading, not compiled, not run on hardware
//
#include <hip/hip_runtime.h>
#include <hip/hip_bf16.h>

// MI455X / gfx1250 multi-head attention, bf16 WMMA pipeline.
// B=4, S=2048, D=1024, H=16, DK=64.

typedef __bf16 bf16_t;
typedef __attribute__((ext_vector_type(16))) __bf16 v16bf;
typedef __attribute__((ext_vector_type(8)))  __bf16 v8bf;
typedef __attribute__((ext_vector_type(8)))  float  v8f;

static __device__ __forceinline__ v8f zero8() {
    v8f z;
#pragma unroll
    for (int i = 0; i < 8; ++i) z[i] = 0.0f;
    return z;
}

static __device__ __forceinline__ v8f wmma_bf16(v16bf a, v16bf b, v8f c) {
    // D = A(16x32 bf16) x B(32x16 bf16) + C(16x16 f32)
    return __builtin_amdgcn_wmma_f32_16x16x32_bf16(
        /*neg_a=*/false, a, /*neg_b=*/false, b,
        /*c_mod=*/(short)0, c, /*reuse_a=*/false, /*reuse_b=*/false);
}

// gfx1250 async memory->LDS copy (ASYNCcnt-tracked, no VGPR round trip).
// Low 32 bits of a generic pointer to __shared__ are the LDS byte offset
// (ISA flat addressing).
static __device__ __forceinline__ void async_copy_b128(void* lds_dst,
                                                       const void* gsrc) {
    unsigned int l = (unsigned int)(unsigned long long)lds_dst;
    asm volatile("global_load_async_to_lds_b128 %0, %1, off"
                 :: "v"(l), "v"(gsrc) : "memory");
}
static __device__ __forceinline__ void wait_async0() {
    asm volatile("s_wait_asynccnt 0x0" ::: "memory");
}

// ---------------------------------------------------------------------------
// Kernel 1: f32 -> bf16 weight conversion (1M elems per matrix)
// ---------------------------------------------------------------------------
__global__ void cvt_f32_to_bf16(const float* __restrict__ src,
                                bf16_t* __restrict__ dst, int n) {
    int i = (blockIdx.x * blockDim.x + threadIdx.x) * 8;
    if (i >= n) return;
    v8f x = *(const v8f*)(src + i);
    v8bf y;
#pragma unroll
    for (int j = 0; j < 8; ++j) y[j] = (bf16_t)x[j];
    *(v8bf*)(dst + i) = y;
}

// ---------------------------------------------------------------------------
// Kernel 2: projection GEMM  Y[m,e] = (sum_d X[m,d]*W[e,d] + b[e]) * scale
// X: [8192 x 1024] f32 (converted in-flight), W: [1024 x 1024] bf16 row-major.
// Output written as bf16 in [B, H, S, DK] layout.
// Block = 256 thr (8 waves); block tile 256(M) x 64(N); wave tile 32 x 64
// (two A fragments reuse each B fragment -> 8 WMMA per 4 B loads per k-step).
// ---------------------------------------------------------------------------
__global__ __launch_bounds__(256)
void qkv_proj_kernel(const float* __restrict__ X,
                     const bf16_t* __restrict__ Wb,
                     const float* __restrict__ bias,
                     bf16_t* __restrict__ Out,
                     float scale) {
    const int lane = threadIdx.x & 31;
    const int wave = threadIdx.x >> 5;
    const int half = lane >> 4;
    const int l16  = lane & 15;
    const int m0 = blockIdx.x * 256 + wave * 32;
    const int n0 = blockIdx.y * 64;
    const float* xrow0 = X + (size_t)(m0 + l16) * 1024;
    const float* xrow1 = X + (size_t)(m0 + 16 + l16) * 1024;

    v8f acc[2][4];
#pragma unroll
    for (int r = 0; r < 2; ++r)
#pragma unroll
        for (int t = 0; t < 4; ++t) acc[r][t] = zero8();

    for (int k0 = 0; k0 < 1024; k0 += 32) {
        // A fragments: elems 0..15 -> d=k0+16*half+i
        v8f lo0 = *(const v8f*)(xrow0 + k0 + 16 * half);
        v8f hi0 = *(const v8f*)(xrow0 + k0 + 16 * half + 8);
        v8f lo1 = *(const v8f*)(xrow1 + k0 + 16 * half);
        v8f hi1 = *(const v8f*)(xrow1 + k0 + 16 * half + 8);
        v16bf a0, a1;
#pragma unroll
        for (int j = 0; j < 8; ++j) {
            a0[j] = (bf16_t)lo0[j]; a0[j + 8] = (bf16_t)hi0[j];
            a1[j] = (bf16_t)lo1[j]; a1[j + 8] = (bf16_t)hi1[j];
        }
        // hint activations two k-blocks ahead into cache (global_prefetch_b8)
        __builtin_prefetch(xrow0 + k0 + 64, 0, 1);
        __builtin_prefetch(xrow1 + k0 + 64, 0, 1);
#pragma unroll
        for (int t = 0; t < 4; ++t) {
            // B fragment: B[d][e] = W[e][d]; n = e = n0+16t+l16, d = k0+16*half+i
            const bf16_t* wrow = Wb + (size_t)(n0 + 16 * t + l16) * 1024;
            v16bf b = *(const v16bf*)(wrow + k0 + 16 * half);
            acc[0][t] = wmma_bf16(a0, b, acc[0][t]);
            acc[1][t] = wmma_bf16(a1, b, acc[1][t]);
        }
    }

#pragma unroll
    for (int t = 0; t < 4; ++t) {
        const int e  = n0 + 16 * t + l16;
        const float bv = bias[e];
        const int h  = e >> 6;
        const int dk = e & 63;
#pragma unroll
        for (int r = 0; r < 2; ++r)
#pragma unroll
            for (int i = 0; i < 8; ++i) {
                int mr   = m0 + 16 * r + i + 8 * half;  // global row = b*S + s
                int bidx = mr >> 11;
                int s    = mr & 2047;
                float y  = (acc[r][t][i] + bv) * scale;
                Out[(((size_t)bidx * 16 + h) * 2048 + s) * 64 + dk] = (bf16_t)y;
            }
    }
}

// ---------------------------------------------------------------------------
// Kernel 3: flash attention per (b,h). Q pre-scaled by 1/sqrt(DK).
// Block = 256 thr (8 waves) covers 128 query rows; key loop in tiles of 32.
// K tile [32x64] staged via GLOBAL_LOAD_ASYNC_TO_LDS_B128 (ASYNCcnt path);
// V tile transposed [64x32] staged through registers; P transposed through
// per-wave LDS scratch to build WMMA A-fragments.
// ---------------------------------------------------------------------------
__global__ __launch_bounds__(256)
void flash_attn_kernel(const bf16_t* __restrict__ Qb,
                       const bf16_t* __restrict__ Kb,
                       const bf16_t* __restrict__ Vb,
                       const int* __restrict__ mask,
                       bf16_t* __restrict__ O) {
    __shared__ alignas(64) bf16_t Kls[32 * 64];     // [key][d]
    __shared__ alignas(64) bf16_t Vls[64 * 32];     // [d][key] (transposed)
    __shared__ alignas(64) bf16_t Pls[8 * 16 * 32]; // per-wave 16x32 P scratch

    const int lane = threadIdx.x & 31;
    const int wave = threadIdx.x >> 5;
    const int half = lane >> 4;
    const int l16  = lane & 15;
    const int b = blockIdx.z, h = blockIdx.y;
    const int bh = b * 16 + h;
    const int q0 = blockIdx.x * 128 + wave * 16;

    const bf16_t* Qp = Qb + (size_t)bh * 2048 * 64;
    const bf16_t* Kp = Kb + (size_t)bh * 2048 * 64;
    const bf16_t* Vp = Vb + (size_t)bh * 2048 * 64;

    // Q A-fragments (held in registers for whole kernel): rows q0+l16
    const bf16_t* qrow = Qp + (size_t)(q0 + l16) * 64;
    v16bf aQ[2];
#pragma unroll
    for (int f = 0; f < 2; ++f) {
        v8bf lo = *(const v8bf*)(qrow + 32 * f + 16 * half);
        v8bf hi = *(const v8bf*)(qrow + 32 * f + 16 * half + 8);
#pragma unroll
        for (int j = 0; j < 8; ++j) { aQ[f][j] = lo[j]; aQ[f][j + 8] = hi[j]; }
    }

    float mrun[8], lrun[8];
    v8f accO[4];
#pragma unroll
    for (int i = 0; i < 8; ++i) { mrun[i] = -__builtin_inff(); lrun[i] = 0.0f; }
#pragma unroll
    for (int t = 0; t < 4; ++t) accO[t] = zero8();

    for (int kt = 0; kt < 2048; kt += 32) {
        __syncthreads();
        {   // stage K (async direct-to-LDS) and V (transposed via registers)
            int row = threadIdx.x >> 3;        // 0..31 key within tile
            int col = (threadIdx.x & 7) * 8;   // 0..56 d
            async_copy_b128((char*)Kls + threadIdx.x * 16,
                            Kp + (size_t)(kt + row) * 64 + col);
            v8bf vv = *(const v8bf*)(Vp + (size_t)(kt + row) * 64 + col);
#pragma unroll
            for (int j = 0; j < 8; ++j) Vls[(col + j) * 32 + row] = vv[j];
            wait_async0();
        }
        __syncthreads();

        // scores: two 16-key tiles, contraction over d=64 in two WMMA steps
        v8f sc[2];
#pragma unroll
        for (int st = 0; st < 2; ++st) {
            int kk = l16 + 16 * st;
            v16bf b0 = *(const v16bf*)(Kls + kk * 64 + 16 * half);
            v16bf b1 = *(const v16bf*)(Kls + kk * 64 + 32 + 16 * half);
            v8f z = zero8();
            z = wmma_bf16(aQ[0], b0, z);
            z = wmma_bf16(aQ[1], b1, z);
            sc[st] = z;
        }

        // mask + per-row max (row m = i + 8*half lives in one 16-lane group)
        float tmax[8];
#pragma unroll
        for (int i = 0; i < 8; ++i) {
            int q = q0 + i + 8 * half;
            const int* mq = mask + ((size_t)b * 2048 + q) * 2048 + kt;
            float s0 = mq[l16]      ? sc[0][i] : -1.0e9f;
            float s1 = mq[16 + l16] ? sc[1][i] : -1.0e9f;
            sc[0][i] = s0; sc[1][i] = s1;
            tmax[i] = fmaxf(s0, s1);
        }
#pragma unroll
        for (int i = 0; i < 8; ++i)
#pragma unroll
            for (int d = 1; d < 16; d <<= 1)
                tmax[i] = fmaxf(tmax[i], __shfl_xor(tmax[i], d, 32));

        // online softmax update
        float alpha[8], rsum[8];
#pragma unroll
        for (int i = 0; i < 8; ++i) {
            float mnew = fmaxf(mrun[i], tmax[i]);
            alpha[i] = __expf(mrun[i] - mnew);
            float p0 = __expf(sc[0][i] - mnew);
            float p1 = __expf(sc[1][i] - mnew);
            sc[0][i] = p0; sc[1][i] = p1;
            rsum[i] = p0 + p1;
            mrun[i] = mnew;
        }
#pragma unroll
        for (int i = 0; i < 8; ++i)
#pragma unroll
            for (int d = 1; d < 16; d <<= 1)
                rsum[i] += __shfl_xor(rsum[i], d, 32);
#pragma unroll
        for (int i = 0; i < 8; ++i) lrun[i] = lrun[i] * alpha[i] + rsum[i];
#pragma unroll
        for (int t = 0; t < 4; ++t)
#pragma unroll
            for (int i = 0; i < 8; ++i) accO[t][i] *= alpha[i];

        // transpose P (C-layout) -> A-fragment via per-wave LDS scratch
        bf16_t* pw = Pls + wave * 512;
#pragma unroll
        for (int st = 0; st < 2; ++st)
#pragma unroll
            for (int i = 0; i < 8; ++i)
                pw[(i + 8 * half) * 32 + (l16 + 16 * st)] = (bf16_t)sc[st][i];
        // wave-private region: LDS ops from one wave stay in order
        v16bf aP;
        {
            const bf16_t* pr = pw + l16 * 32;
            v8bf plo = *(const v8bf*)(pr + 16 * half);
            v8bf phi = *(const v8bf*)(pr + 16 * half + 8);
#pragma unroll
            for (int j = 0; j < 8; ++j) { aP[j] = plo[j]; aP[j + 8] = phi[j]; }
        }

        // O += P(16x32) x V(32x64): B from transposed V, contiguous per lane
#pragma unroll
        for (int t = 0; t < 4; ++t) {
            v16bf bv = *(const v16bf*)(Vls + (16 * t + l16) * 32 + 16 * half);
            accO[t] = wmma_bf16(aP, bv, accO[t]);
        }
    }

    // epilogue: normalize and store to [B, S, D] bf16 for the output GEMM
    float inv[8];
#pragma unroll
    for (int i = 0; i < 8; ++i) inv[i] = 1.0f / lrun[i];
#pragma unroll
    for (int t = 0; t < 4; ++t) {
        int d = 16 * t + l16;
#pragma unroll
        for (int i = 0; i < 8; ++i) {
            int q = q0 + i + 8 * half;
            O[((size_t)b * 2048 + q) * 1024 + h * 64 + d] =
                (bf16_t)(accO[t][i] * inv[i]);
        }
    }
}

// ---------------------------------------------------------------------------
// Kernel 4: output projection, bf16 A x bf16 Wo^T -> f32 out + bias.
// Same 256x64 block tile / 32x64 wave tile as kernel 2.
// ---------------------------------------------------------------------------
__global__ __launch_bounds__(256)
void out_proj_kernel(const bf16_t* __restrict__ A,
                     const bf16_t* __restrict__ Wb,
                     const float* __restrict__ bias,
                     float* __restrict__ Out) {
    const int lane = threadIdx.x & 31;
    const int wave = threadIdx.x >> 5;
    const int half = lane >> 4;
    const int l16  = lane & 15;
    const int m0 = blockIdx.x * 256 + wave * 32;
    const int n0 = blockIdx.y * 64;
    const bf16_t* arow0 = A + (size_t)(m0 + l16) * 1024;
    const bf16_t* arow1 = A + (size_t)(m0 + 16 + l16) * 1024;

    v8f acc[2][4];
#pragma unroll
    for (int r = 0; r < 2; ++r)
#pragma unroll
        for (int t = 0; t < 4; ++t) acc[r][t] = zero8();

    for (int k0 = 0; k0 < 1024; k0 += 32) {
        v8bf lo0 = *(const v8bf*)(arow0 + k0 + 16 * half);
        v8bf hi0 = *(const v8bf*)(arow0 + k0 + 16 * half + 8);
        v8bf lo1 = *(const v8bf*)(arow1 + k0 + 16 * half);
        v8bf hi1 = *(const v8bf*)(arow1 + k0 + 16 * half + 8);
        v16bf a0, a1;
#pragma unroll
        for (int j = 0; j < 8; ++j) {
            a0[j] = lo0[j]; a0[j + 8] = hi0[j];
            a1[j] = lo1[j]; a1[j + 8] = hi1[j];
        }
        __builtin_prefetch(arow0 + k0 + 128, 0, 1);
        __builtin_prefetch(arow1 + k0 + 128, 0, 1);
#pragma unroll
        for (int t = 0; t < 4; ++t) {
            const bf16_t* wrow = Wb + (size_t)(n0 + 16 * t + l16) * 1024;
            v16bf b = *(const v16bf*)(wrow + k0 + 16 * half);
            acc[0][t] = wmma_bf16(a0, b, acc[0][t]);
            acc[1][t] = wmma_bf16(a1, b, acc[1][t]);
        }
    }

#pragma unroll
    for (int t = 0; t < 4; ++t) {
        const int e = n0 + 16 * t + l16;
        const float bv = bias[e];
#pragma unroll
        for (int r = 0; r < 2; ++r)
#pragma unroll
            for (int i = 0; i < 8; ++i) {
                int mr = m0 + 16 * r + i + 8 * half;
                Out[(size_t)mr * 1024 + e] = acc[r][t][i] + bv;
            }
    }
}

// ---------------------------------------------------------------------------
extern "C" void kernel_launch(void* const* d_in, const int* in_sizes, int n_in,
                              void* d_out, int out_size, void* d_ws, size_t ws_size,
                              hipStream_t stream) {
    const float* q   = (const float*)d_in[0];
    const float* k   = (const float*)d_in[1];
    const float* v   = (const float*)d_in[2];
    const int*   msk = (const int*)  d_in[3];
    const float* wq  = (const float*)d_in[4];
    const float* bq  = (const float*)d_in[5];
    const float* wk  = (const float*)d_in[6];
    const float* bk  = (const float*)d_in[7];
    const float* wv  = (const float*)d_in[8];
    const float* bv  = (const float*)d_in[9];
    const float* wo  = (const float*)d_in[10];
    const float* bo  = (const float*)d_in[11];
    float* out = (float*)d_out;

    // workspace layout (bf16 elements): 4 weights (1M ea) + Q/K/V (8M ea) + O (8M)
    bf16_t* ws  = (bf16_t*)d_ws;
    const size_t MW = 1024u * 1024u;
    bf16_t* wqb = ws;
    bf16_t* wkb = ws + 1 * MW;
    bf16_t* wvb = ws + 2 * MW;
    bf16_t* wob = ws + 3 * MW;
    bf16_t* Qb  = ws + 4 * MW;
    bf16_t* Kb  = Qb + 8 * MW;
    bf16_t* Vb  = Kb + 8 * MW;
    bf16_t* Ob  = Vb + 8 * MW;

    dim3 cblk(256);
    dim3 cgrd((unsigned)(MW / (256 * 8)));
    cvt_f32_to_bf16<<<cgrd, cblk, 0, stream>>>(wq, wqb, (int)MW);
    cvt_f32_to_bf16<<<cgrd, cblk, 0, stream>>>(wk, wkb, (int)MW);
    cvt_f32_to_bf16<<<cgrd, cblk, 0, stream>>>(wv, wvb, (int)MW);
    cvt_f32_to_bf16<<<cgrd, cblk, 0, stream>>>(wo, wob, (int)MW);

    dim3 pg(32, 16, 1), pb(256);
    qkv_proj_kernel<<<pg, pb, 0, stream>>>(q, wqb, bq, Qb, 0.125f); // fold 1/sqrt(DK)
    qkv_proj_kernel<<<pg, pb, 0, stream>>>(k, wkb, bk, Kb, 1.0f);
    qkv_proj_kernel<<<pg, pb, 0, stream>>>(v, wvb, bv, Vb, 1.0f);

    dim3 ag(16, 16, 4); // S/128 q-blocks, H, B
    flash_attn_kernel<<<ag, pb, 0, stream>>>(Qb, Kb, Vb, msk, Ob);

    out_proj_kernel<<<pg, pb, 0, stream>>>(Ob, wob, bo, out);
}
